// AttentionHead_34213709480279
// MI455X (gfx1250) — matrix operand order repaired and checked by reading, not compiled, not run on hardware
//
#include <hip/hip_runtime.h>
#include <math.h>

typedef __attribute__((ext_vector_type(16))) _Float16 v16h;
typedef __attribute__((ext_vector_type(8)))  float    v8f;

#define WMMA_F16(A, B, C) \
    __builtin_amdgcn_wmma_f32_16x16x32_f16(false, (A), false, (B), (short)0, (C), false, false)

// Problem constants
#define BB 8
#define SS 2048
#define DD 1024
#define HH 64

// ---------------------------------------------------------------------------
// Async 16-byte global -> LDS copy (per-lane), tracked by ASYNCcnt.
// VDST operand = LDS byte offset in a VGPR; generic pointers to __shared__
// carry the workgroup-relative LDS offset in their low 32 bits.
// ---------------------------------------------------------------------------
__device__ __forceinline__ void async_copy16(void* lds_dst, const void* gsrc) {
    unsigned loff = (unsigned)(unsigned long long)(lds_dst);
    asm volatile("global_load_async_to_lds_b128 %0, %1, off"
                 :: "v"(loff), "v"(gsrc) : "memory");
}

__device__ __forceinline__ void wait_async0() {
#if __has_builtin(__builtin_amdgcn_s_wait_asynccnt)
    __builtin_amdgcn_s_wait_asynccnt(0);
#else
    asm volatile("s_wait_asynccnt 0" ::: "memory");
#endif
}

// ---------------------------------------------------------------------------
// Pass 0: convert W (fp32 [D,H] x3) into f16 transposed, fragment-ready:
//   wt[(mat*64 + col)*1024 + k] = Wmat[k*64 + col]
// 192*1024 elements.
// ---------------------------------------------------------------------------
__global__ __launch_bounds__(256) void wcvt_kernel(
    const float* __restrict__ Wq, const float* __restrict__ Wk,
    const float* __restrict__ Wv, _Float16* __restrict__ wt)
{
    const int idx = blockIdx.x * 256 + threadIdx.x;
    if (idx >= 192 * 1024) return;
    const int gcol = idx >> 10;      // 0..191
    const int k    = idx & 1023;
    const int mat  = gcol >> 6;
    const int col  = gcol & 63;
    const float* wm = (mat == 0) ? Wq : (mat == 1) ? Wk : Wv;
    wt[(size_t)gcol * 1024 + k] = (_Float16)wm[(size_t)k * HH + col];
}

// ---------------------------------------------------------------------------
// Pass 1: fused QKV projection with async double-buffered W staging.
//   Q,K -> f16 [B*S, H]; V -> f16 transposed [B, H, S].
// 8 waves/block; wave = one 16-row M tile, all 192 N columns.
// ---------------------------------------------------------------------------
__global__ __launch_bounds__(256) void qkv_proj_kernel(
    const float* __restrict__ x,
    const float* __restrict__ bq, const float* __restrict__ bk,
    const float* __restrict__ bv, const _Float16* __restrict__ wt,
    _Float16* __restrict__ qh, _Float16* __restrict__ kh,
    _Float16* __restrict__ vt)
{
    // lw[buf][gcol][kk 0..31], fragment-ready (contiguous 16-half B reads)
    __shared__ __align__(32) _Float16 lw[2][192 * 32];   // 24 KB

    const int tid  = threadIdx.x;
    const int wave = tid >> 5;
    const int lane = tid & 31;
    const int hi   = lane >> 4;
    const int lo   = lane & 15;
    const int mbase = blockIdx.x * 128 + wave * 16;
    const int arow  = mbase + lo;

    // Stage one 32-deep k-chunk of all 192 W columns: 768 x 16B, 3 per thread.
    auto stage_w = [&](int buf, int k0) {
        #pragma unroll
        for (int i = 0; i < 3; i++) {
            const int c    = tid + i * 256;
            const int gcol = c >> 2;
            const int part = c & 3;
            async_copy16(&lw[buf][gcol * 32 + part * 8],
                         wt + (size_t)gcol * 1024 + k0 + part * 8);
        }
    };

    v8f c[12];
    #pragma unroll
    for (int i = 0; i < 12; i++) {
        #pragma unroll
        for (int r = 0; r < 8; r++) c[i][r] = 0.0f;
    }

    stage_w(0, 0);   // prologue

    for (int ks = 0; ks < 32; ks++) {
        const int k0 = ks * 32;
        wait_async0();        // our chunk landed
        __syncthreads();      // everyone's chunk landed; prev buffer reads done
        if (ks < 31) stage_w((ks + 1) & 1, k0 + 32);   // overlap next copy

        // ---- A fragment (16x32 f16) from x, fp32->f16 in regs
        const float* xr = x + (size_t)arow * DD + k0 + hi * 16;
        if (k0 + 32 < DD) __builtin_prefetch(xr + 32, 0, 0);
        v16h a;
        #pragma unroll
        for (int j = 0; j < 8; j++) {
            a[j]     = (_Float16)xr[j];
            a[8 + j] = (_Float16)xr[8 + j];
        }

        // ---- 12 B fragments from LDS, software-pipelined against WMMA
        const _Float16* buf = lw[ks & 1];
        v16h bnext = *(const v16h*)(buf + (0 * 16 + lo) * 32 + hi * 16);
        #pragma unroll
        for (int nt = 0; nt < 12; nt++) {
            v16h bcur = bnext;
            if (nt < 11)
                bnext = *(const v16h*)(buf + ((nt + 1) * 16 + lo) * 32 + hi * 16);
            c[nt] = WMMA_F16(a, bcur, c[nt]);
        }
    }

    // ---- bias + store. C layout: (M = r + 8*hi, N = lo)
    #pragma unroll
    for (int nt = 0; nt < 12; nt++) {
        const int mat  = nt >> 2;
        const int colm = (nt & 3) * 16 + lo;
        const float* bias = (mat == 0) ? bq : (mat == 1) ? bk : bv;
        const float bb = bias[colm];
        #pragma unroll
        for (int r = 0; r < 8; r++) {
            const int gm  = mbase + r + 8 * hi;
            const float v = c[nt][r] + bb;
            if (mat == 0) {
                qh[(size_t)gm * HH + colm] = (_Float16)v;
            } else if (mat == 1) {
                kh[(size_t)gm * HH + colm] = (_Float16)v;
            } else {
                const int bidx = gm >> 11;
                const int s    = gm & (SS - 1);
                vt[((size_t)bidx * HH + colm) * SS + s] = (_Float16)v;
            }
        }
    }
}

// ---------------------------------------------------------------------------
// Pass 2: causal flash attention.
// Block = 8 waves = 128 consecutive q rows (one batch). All waves walk the
// block-max number of 32-key blocks; K/V tiles are cooperatively async-staged
// into double-buffered LDS (8x traffic reduction); waves past their causal
// window skip compute (wave-uniform branch -> EXEC stays all-1s for WMMA).
// ---------------------------------------------------------------------------
__global__ __launch_bounds__(256) void flash_attn_kernel(
    const _Float16* __restrict__ qh, const _Float16* __restrict__ kh,
    const _Float16* __restrict__ vt, float* __restrict__ out)
{
    __shared__ __align__(32) _Float16 lk[2][32 * 64];   // K tile  [key][h]  8 KB
    __shared__ __align__(32) _Float16 lv[2][64 * 32];   // V tile  [h][key]  8 KB
    __shared__ __align__(32) _Float16 sp[8][16 * 32];   // per-wave P        8 KB

    const int tid  = threadIdx.x;
    const int wave = tid >> 5;
    const int lane = tid & 31;
    const int hi   = lane >> 4;
    const int lo   = lane & 15;

    const int qbase = blockIdx.x * 128 + wave * 16;     // global query row
    const int bidx  = qbase >> 11;
    const int qseq  = qbase & (SS - 1);
    const int qseq0 = (blockIdx.x * 128) & (SS - 1);
    const int nkb_max = ((qseq0 + 127) >> 5) + 1;       // uniform per block
    _Float16* myp = sp[wave];

    // Cooperative staging of one 32-key K/V tile: 512 x 16B, 2 per thread.
    auto stage_kv = [&](int buf, int kbase) {
        {   // K: 32 keys x 64 h  (rows of kh)
            const int key = tid >> 3, part = tid & 7;
            async_copy16(&lk[buf][key * 64 + part * 8],
                         kh + ((size_t)bidx * SS + kbase + key) * HH + part * 8);
        }
        {   // V: 64 h x 32 keys  (rows of vt)
            const int h = tid >> 2, part = tid & 3;
            async_copy16(&lv[buf][h * 32 + part * 8],
                         vt + ((size_t)bidx * HH + h) * SS + kbase + part * 8);
        }
    };

    // ---- Q A-fragments (H=64 -> two 16x32 chunks)
    v16h aq0, aq1;
    {
        const _Float16* qr = qh + (size_t)(qbase + lo) * HH + hi * 16;
        #pragma unroll
        for (int j = 0; j < 8; j++) {
            aq0[j]     = qr[j];
            aq0[8 + j] = qr[8 + j];
            aq1[j]     = qr[32 + j];
            aq1[8 + j] = qr[40 + j];
        }
    }

    v8f acc[4];
    #pragma unroll
    for (int i = 0; i < 4; i++) {
        #pragma unroll
        for (int r = 0; r < 8; r++) acc[i][r] = 0.0f;
    }
    float mi[8], li[8];
    #pragma unroll
    for (int r = 0; r < 8; r++) { mi[r] = -INFINITY; li[r] = 0.0f; }

    stage_kv(0, 0);   // prologue

    for (int kb = 0; kb < nkb_max; kb++) {
        const int kbase = kb * 32;
        wait_async0();
        __syncthreads();
        if (kb + 1 < nkb_max) stage_kv((kb + 1) & 1, kbase + 32);

        if (kbase <= qseq + 15) {            // wave-uniform causal guard
            const _Float16* tk = lk[kb & 1];
            const _Float16* tv = lv[kb & 1];

            // ---- scores S = Q @ K^T : two 16x16 tiles
            v8f s[2];
            #pragma unroll
            for (int t = 0; t < 2; t++) {
                const _Float16* kr = tk + (t * 16 + lo) * 64 + hi * 16;
                v16h f0 = *(const v16h*)(kr);
                v16h f1 = *(const v16h*)(kr + 32);
                v8f z;
                #pragma unroll
                for (int r = 0; r < 8; r++) z[r] = 0.0f;
                z = WMMA_F16(aq0, f0, z);
                z = WMMA_F16(aq1, f1, z);
                s[t] = z;
            }

            // ---- scale + causal mask (C layout: M = r+8*hi, N = lo)
            #pragma unroll
            for (int t = 0; t < 2; t++) {
                #pragma unroll
                for (int r = 0; r < 8; r++) {
                    const int qi = qseq + r + 8 * hi;
                    const int ki = kbase + t * 16 + lo;
                    const float vsc = s[t][r] * 0.125f;   // 1/sqrt(64)
                    s[t][r] = (ki <= qi) ? vsc : -INFINITY;
                }
            }

            // ---- online softmax (rows live in 16-lane halves)
            #pragma unroll
            for (int r = 0; r < 8; r++) {
                float mrow = fmaxf(s[0][r], s[1][r]);
                #pragma unroll
                for (int off = 8; off >= 1; off >>= 1)
                    mrow = fmaxf(mrow, __shfl_xor(mrow, off, 32));
                const float mn    = fmaxf(mi[r], mrow);
                const float alpha = __expf(mi[r] - mn);
                const float p0    = __expf(s[0][r] - mn);
                const float p1    = __expf(s[1][r] - mn);
                float rs = p0 + p1;
                #pragma unroll
                for (int off = 8; off >= 1; off >>= 1)
                    rs += __shfl_xor(rs, off, 32);
                li[r] = li[r] * alpha + rs;
                mi[r] = mn;
                #pragma unroll
                for (int nt = 0; nt < 4; nt++) acc[nt][r] *= alpha;
                myp[(r + 8 * hi) * 32 + lo]      = (_Float16)p0;
                myp[(r + 8 * hi) * 32 + 16 + lo] = (_Float16)p1;
            }
            asm volatile("s_wait_dscnt 0" ::: "memory");

            // ---- P as A fragment (16x32)
            v16h ap;
            {
                const _Float16* pr = myp + lo * 32 + hi * 16;
                #pragma unroll
                for (int j = 0; j < 8; j++) { ap[j] = pr[j]; ap[8 + j] = pr[8 + j]; }
            }

            // ---- O += P @ V
            #pragma unroll
            for (int nt = 0; nt < 4; nt++) {
                v16h vf = *(const v16h*)(tv + (nt * 16 + lo) * 32 + hi * 16);
                acc[nt] = WMMA_F16(ap, vf, acc[nt]);
            }
        }
    }

    // ---- epilogue: out = acc / l  (fp32)
    #pragma unroll
    for (int nt = 0; nt < 4; nt++) {
        #pragma unroll
        for (int r = 0; r < 8; r++) {
            const int gm = qbase + r + 8 * hi;
            out[(size_t)gm * HH + nt * 16 + lo] = acc[nt][r] / li[r];
        }
    }
}

// ---------------------------------------------------------------------------
extern "C" void kernel_launch(void* const* d_in, const int* in_sizes, int n_in,
                              void* d_out, int out_size, void* d_ws, size_t ws_size,
                              hipStream_t stream)
{
    const float* x  = (const float*)d_in[0];
    const float* Wq = (const float*)d_in[1];
    const float* bq = (const float*)d_in[2];
    const float* Wk = (const float*)d_in[3];
    const float* bk = (const float*)d_in[4];
    const float* Wv = (const float*)d_in[5];
    const float* bv = (const float*)d_in[6];

    const size_t NE = (size_t)BB * SS * HH;      // 1M
    _Float16* qh = (_Float16*)d_ws;
    _Float16* kh = qh + NE;
    _Float16* vt = kh + NE;                      // [B, H, S]
    _Float16* wt = vt + NE;                      // [192, 1024] f16 (384 KB)

    // Pass 0: W -> f16 transposed fragment-ready layout
    wcvt_kernel<<<dim3((192 * 1024 + 255) / 256), dim3(256), 0, stream>>>(
        Wq, Wk, Wv, wt);

    // Pass 1: QKV projection
    qkv_proj_kernel<<<dim3((BB * SS) / 128), dim3(256), 0, stream>>>(
        x, bq, bk, bv, wt, qh, kh, vt);

    // Pass 2: causal flash attention
    flash_attn_kernel<<<dim3((BB * SS) / 128), dim3(256), 0, stream>>>(
        qh, kh, vt, (float*)d_out);
}
